// CrossAttentionModule_26482768347482
// MI455X (gfx1250) — matrix operand order repaired and checked by reading, not compiled, not run on hardware
//
#include <hip/hip_runtime.h>
#include <hip/hip_bf16.h>
#include <math.h>

// ---------------------------------------------------------------------------
// CDNA5 (gfx1250) fused cross-attention block, round 3.
//  - wave32 WMMA bf16 (v_wmma_f32_16x16x32_bf16) for all four GEMMs
//  - weights pre-packed to bf16 WMMA B-fragment layout in workspace
//  - M=64 rows/block: one B-fragment feeds 4 A-fragments (4 m-tile accums)
//    -> weight L2 traffic halved again vs round 2 (~1.3 GB total)
//  - FFN hidden dim processed in 4 chunks of 256; FFN2 output accumulated in
//    persistent per-wave registers across chunks (no 64x1024 LDS tile)
//  - dynamic LDS ~138 KB/block; 320 KB/WGP -> 2 blocks/WGP, 16 waves
// ---------------------------------------------------------------------------

typedef __bf16 bf16;
typedef __attribute__((ext_vector_type(16))) __bf16 v16bf;
typedef __attribute__((ext_vector_type(8)))  __bf16 v8bf;
typedef __attribute__((ext_vector_type(8)))  float  v8f;

#define LDX  264   // f32 residual tile stride  (row = 1056 B, 16B aligned)
#define LDA  272   // bf16 LN/ctx tile stride   (row = 544 B, 16B aligned)
#define LDHC 272   // bf16 hidden-chunk/Q tile stride

#define M_ROWS 64  // rows per block (4 WMMA m-tiles)
#define MT     4   // m-tiles per block
#define CHUNK  256 // FFN hidden columns per chunk
#define NCHUNK 4   // FF / CHUNK

// LDS carve sizes (bytes)
#define XF_BYTES (M_ROWS * LDX * 4)        // 67584
#define NB_BYTES (M_ROWS * LDA * 2)        // 34816
#define HB_BYTES (M_ROWS * LDHC * 2)       // 34816
#define AT_BYTES (M_ROWS * 8 * 2 * 4)      // 4096
#define SMEM_BYTES (XF_BYTES + NB_BYTES + HB_BYTES + AT_BYTES)  // 141312

// ---------------------------------------------------------------------------
// A fragment (16x32 bf16) from LDS, row-major with stride lda.
// ISA 7.12.2 16-bit A layout: lane m = lane&15; lanes 0-15 hold K 0-7/16-23,
// lanes 16-31 hold K 8-15/24-31; each lane's data = two contiguous 8-elem runs.
// ---------------------------------------------------------------------------
__device__ __forceinline__ v16bf load_a_bf(const bf16* A, int lda, int mbase,
                                           int kbase) {
  int lane = threadIdx.x & 31;
  const bf16* row = A + (mbase + (lane & 15)) * lda + kbase + (lane >> 4) * 8;
  v8bf lo = *(const v8bf*)(row);        // K = kbase+half*8    .. +7
  v8bf hi = *(const v8bf*)(row + 16);   // K = kbase+16+half*8 .. +7
  return __builtin_shufflevector(lo, hi, 0, 1, 2, 3, 4, 5, 6, 7,
                                 8, 9, 10, 11, 12, 13, 14, 15);
}

// ---------------------------------------------------------------------------
// Packed-weight B fragment: fragment (t, kb) = 32 lanes x 16 bf16 contiguous,
// exactly the ISA B layout (lane n = lane&15, lanes 0-15 hold K 0-7/16-23, lanes 16-31
// K 8-15/24-31). One aligned 32-byte load per lane.
// ---------------------------------------------------------------------------
__device__ __forceinline__ v16bf load_b_packed(const bf16* __restrict__ P,
                                               int nk, int t, int kb) {
  int lane = threadIdx.x & 31;
  return *(const v16bf*)(P + ((long)(t * nk + kb) * 32 + lane) * 16);
}

__device__ __forceinline__ v8f wmma_bf16(v16bf a, v16bf b, v8f c) {
  return __builtin_amdgcn_wmma_f32_16x16x32_bf16(
      /*neg_a=*/false, a, /*neg_b=*/false, b,
      /*c_mod=*/(short)0, c, /*reuse_a=*/false, /*reuse_b=*/false);
}

__device__ __forceinline__ float gelu_exact(float x) {
  return 0.5f * x * (1.0f + erff(x * 0.70710678118654752f));
}

// ---------------------------------------------------------------------------
// Pack one weight matrix W[rows][K] (fp32, row-major, used as X @ W^T) into
// bf16 fragment layout. grid = ntiles*nk blocks of 32 threads.
// dst[((t*nk+kb)*32 + lane)*16 + j] = W[(t*16 + (lane&15))*K
//     + kb*32 + (lane>>4)*8 + (j < 8 ? j : j + 8)]
// ---------------------------------------------------------------------------
__global__ void __launch_bounds__(32) pack_weight_kernel(
    const float* __restrict__ W, bf16* __restrict__ P, int K) {
  int nk = K >> 5;
  int frag = blockIdx.x;  // t*nk + kb
  int lane = threadIdx.x;
  int t = frag / nk;
  int kb = frag - t * nk;
  const float* src =
      W + (long)(t * 16 + (lane & 15)) * K + kb * 32 + (lane >> 4) * 8;
  bf16* dst = P + ((long)frag * 32 + lane) * 16;
#pragma unroll
  for (int j = 0; j < 16; ++j) dst[j] = (bf16)src[j < 8 ? j : j + 8];
}

// ---------------------------------------------------------------------------
// Kernel A: token path.  grid = B*2 blocks of 256 threads.
// proj = tok @ Wp^T + bp ; LN ; K = kvn @ Wk^T + bk ; V = kvn @ Wv^T + bv
// ---------------------------------------------------------------------------
__global__ void __launch_bounds__(256) token_kv_kernel(
    const float* __restrict__ tgt, const float* __restrict__ bg,
    const float* __restrict__ Wp,  const float* __restrict__ bp,
    const float* __restrict__ g_tok, const float* __restrict__ b_tok,
    const float* __restrict__ Wk,  const float* __restrict__ Wv,
    const float* __restrict__ bk,  const float* __restrict__ bv,
    float* __restrict__ Kbuf, float* __restrict__ Vbuf) {
  const int b = blockIdx.x >> 1;
  const int which = blockIdx.x & 1;
  const float* tok = (which == 0 ? tgt : bg) + b * 128;

  __shared__ float kvn[256];
  __shared__ float red[16];

  const int t = threadIdx.x;
  float p = bp[t];
  for (int i = 0; i < 128; ++i) p += tok[i] * Wp[t * 128 + i];

  float s = p, s2 = p * p;
#pragma unroll
  for (int off = 16; off; off >>= 1) {
    s  += __shfl_xor(s, off);
    s2 += __shfl_xor(s2, off);
  }
  int wave = t >> 5, lane = t & 31;
  if (lane == 0) { red[wave] = s; red[8 + wave] = s2; }
  __syncthreads();
  if (t == 0) {
    float ss = 0.f, ss2 = 0.f;
    for (int i = 0; i < 8; ++i) { ss += red[i]; ss2 += red[8 + i]; }
    red[0] = ss * (1.f / 256.f);
    red[1] = ss2 * (1.f / 256.f);
  }
  __syncthreads();
  float mean = red[0];
  float var  = red[1] - mean * mean;
  float rstd = rsqrtf(var + 1e-5f);
  kvn[t] = (p - mean) * rstd * g_tok[t] + b_tok[t];
  __syncthreads();

  float kk = bk[t], vv = bv[t];
  for (int i = 0; i < 256; ++i) {
    float c = kvn[i];
    kk += c * Wk[t * 256 + i];
    vv += c * Wv[t * 256 + i];
  }
  int o = (b * 2 + which) * 256 + t;
  Kbuf[o] = kk;
  Vbuf[o] = vv;
}

// ---------------------------------------------------------------------------
// Kernel B: fused per-64-row-tile block.
// grid = (N/64, B), 256 threads = 8 wave32, dynamic LDS.
// ---------------------------------------------------------------------------
__global__ void __launch_bounds__(256) fused_block_kernel(
    const float* __restrict__ img,
    const float* __restrict__ g_img, const float* __restrict__ b_img,
    const float* __restrict__ g_ffn, const float* __restrict__ b_ffn,
    const bf16* __restrict__ WqP, const float* __restrict__ bq,
    const bf16* __restrict__ WoP, const float* __restrict__ bo,
    const bf16* __restrict__ W1P, const float* __restrict__ b1,
    const bf16* __restrict__ W2P, const float* __restrict__ b2,
    const float* __restrict__ Kbuf, const float* __restrict__ Vbuf,
    float* __restrict__ out) {
  constexpr int N = 4096, E = 256, NH = 8, HD = 32;
  const int b    = blockIdx.y;
  const int n0   = blockIdx.x * M_ROWS;
  const int tid  = threadIdx.x;
  const int wave = tid >> 5;
  const int lane = tid & 31;
  const int half = lane >> 4;
  const int nn   = lane & 15;

  extern __shared__ char smem[];
  float* xf   = (float*)smem;                         // [M_ROWS][LDX]  f32
  bf16*  nbuf = (bf16*)(smem + XF_BYTES);             // [M_ROWS][LDA]  bf16
  bf16*  hbuf = (bf16*)(smem + XF_BYTES + NB_BYTES);  // [M_ROWS][LDHC] bf16
  float* attw = (float*)(smem + XF_BYTES + NB_BYTES + HB_BYTES);  // [M][8][2]

  // ---- 1. load x tile, transposing [B,E,N] -> rows of [N,E] --------------
  for (int i = tid; i < M_ROWS * E; i += 256) {
    int e = i >> 6;          // 0..255 over outer iterations
    int m = i & 63;          // lane-contiguous rows -> coalesced
    xf[m * LDX + e] = img[((long)b * E + e) * N + n0 + m];
  }
  __syncthreads();

  // ---- 2. LayerNorm(image) -> nbuf bf16 ----------------------------------
  for (int r = wave; r < M_ROWS; r += 8) {
    float s = 0.f, s2 = 0.f;
    for (int e = lane; e < E; e += 32) {
      float v = xf[r * LDX + e];
      s += v; s2 += v * v;
    }
#pragma unroll
    for (int off = 16; off; off >>= 1) {
      s  += __shfl_xor(s, off);
      s2 += __shfl_xor(s2, off);
    }
    float mean = s * (1.f / E);
    float var  = s2 * (1.f / E) - mean * mean;
    float rstd = rsqrtf(var + 1e-5f);
    for (int e = lane; e < E; e += 32) {
      float v = (xf[r * LDX + e] - mean) * rstd * g_img[e] + b_img[e];
      nbuf[r * LDA + e] = (bf16)v;
    }
  }
  __syncthreads();

  // ---- 3. Q = LN(x) @ Wq^T + bq -> hbuf bf16 (cols 0..255) ---------------
  for (int t = wave; t < 16; t += 8) {
    v8f acc[MT] = {{}, {}, {}, {}};
#pragma unroll
    for (int kb = 0; kb < 8; ++kb) {
      v16bf bf = load_b_packed(WqP, 8, t, kb);   // loaded once, 4 WMMAs
#pragma unroll
      for (int mt = 0; mt < MT; ++mt)
        acc[mt] = wmma_bf16(load_a_bf(nbuf, LDA, mt * 16, kb * 32), bf, acc[mt]);
    }
    int col = t * 16 + nn;
    float bias = bq[col];
#pragma unroll
    for (int mt = 0; mt < MT; ++mt)
#pragma unroll
      for (int r = 0; r < 8; ++r)
        hbuf[(mt * 16 + r + half * 8) * LDHC + col] = (bf16)(acc[mt][r] + bias);
  }
  __syncthreads();

  // ---- 4. 2-key attention scores + softmax -------------------------------
  for (int p = tid; p < M_ROWS * NH; p += 256) {
    int r = p >> 3;          // 0..63
    int h = p & 7;
    const float* K0 = Kbuf + ((b * 2 + 0) * NH + h) * HD;
    const float* K1 = Kbuf + ((b * 2 + 1) * NH + h) * HD;
    float s0 = 0.f, s1 = 0.f;
    for (int d = 0; d < HD; ++d) {
      float qv = (float)hbuf[r * LDHC + h * HD + d];
      s0 += qv * K0[d];
      s1 += qv * K1[d];
    }
    const float scale = 0.17677669529663687f;  // 1/sqrt(32)
    s0 *= scale; s1 *= scale;
    float mx = fmaxf(s0, s1);
    float e0 = __expf(s0 - mx), e1 = __expf(s1 - mx);
    float inv = 1.f / (e0 + e1);
    attw[(r * NH + h) * 2 + 0] = e0 * inv;
    attw[(r * NH + h) * 2 + 1] = e1 * inv;
  }
  __syncthreads();

  // ---- 5. context = attn @ V -> nbuf bf16 --------------------------------
  for (int i = tid; i < M_ROWS * E; i += 256) {
    int e = i >> 6;
    int m = i & 63;
    int h = e >> 5, d = e & 31;
    float a0 = attw[(m * NH + h) * 2 + 0];
    float a1 = attw[(m * NH + h) * 2 + 1];
    float c = a0 * Vbuf[((b * 2 + 0) * NH + h) * HD + d]
            + a1 * Vbuf[((b * 2 + 1) * NH + h) * HD + d];
    nbuf[m * LDA + e] = (bf16)c;
  }
  __syncthreads();

  // ---- 6. x += ctx @ Wo^T + bo -------------------------------------------
  for (int t = wave; t < 16; t += 8) {
    v8f acc[MT] = {{}, {}, {}, {}};
#pragma unroll
    for (int kb = 0; kb < 8; ++kb) {
      v16bf bf = load_b_packed(WoP, 8, t, kb);
#pragma unroll
      for (int mt = 0; mt < MT; ++mt)
        acc[mt] = wmma_bf16(load_a_bf(nbuf, LDA, mt * 16, kb * 32), bf, acc[mt]);
    }
    int col = t * 16 + nn;
    float bias = bo[col];
#pragma unroll
    for (int mt = 0; mt < MT; ++mt)
#pragma unroll
      for (int r = 0; r < 8; ++r)
        xf[(mt * 16 + r + half * 8) * LDX + col] += acc[mt][r] + bias;
  }
  __syncthreads();

  // ---- 7. LayerNorm(ffn) -> nbuf bf16 ------------------------------------
  for (int r = wave; r < M_ROWS; r += 8) {
    float s = 0.f, s2 = 0.f;
    for (int e = lane; e < E; e += 32) {
      float v = xf[r * LDX + e];
      s += v; s2 += v * v;
    }
#pragma unroll
    for (int off = 16; off; off >>= 1) {
      s  += __shfl_xor(s, off);
      s2 += __shfl_xor(s2, off);
    }
    float mean = s * (1.f / E);
    float var  = s2 * (1.f / E) - mean * mean;
    float rstd = rsqrtf(var + 1e-5f);
    for (int e = lane; e < E; e += 32) {
      float v = (xf[r * LDX + e] - mean) * rstd * g_ffn[e] + b_ffn[e];
      nbuf[r * LDA + e] = (bf16)v;
    }
  }
  __syncthreads();

  // ---- 8/9. FFN in hidden-dim chunks of 256; FFN2 accumulates in regs ----
  // wave w owns output col-tiles {w, w+8}; acc2[i][mt] persists over chunks.
  v8f acc2[2][MT] = {{{}, {}, {}, {}}, {{}, {}, {}, {}}};
  for (int c = 0; c < NCHUNK; ++c) {
    // FFN1 chunk: h[:, c*256 .. c*256+255] = gelu(LN(x) @ W1^T + b1) -> hbuf
    for (int t = wave; t < 16; t += 8) {
      v8f acc[MT] = {{}, {}, {}, {}};
#pragma unroll
      for (int kb = 0; kb < 8; ++kb) {
        v16bf bf = load_b_packed(W1P, 8, c * 16 + t, kb);
#pragma unroll
        for (int mt = 0; mt < MT; ++mt)
          acc[mt] = wmma_bf16(load_a_bf(nbuf, LDA, mt * 16, kb * 32), bf, acc[mt]);
      }
      int cc = t * 16 + nn;             // column within chunk
      float bias = b1[c * CHUNK + cc];
#pragma unroll
      for (int mt = 0; mt < MT; ++mt)
#pragma unroll
        for (int r = 0; r < 8; ++r)
          hbuf[(mt * 16 + r + half * 8) * LDHC + cc] =
              (bf16)gelu_exact(acc[mt][r] + bias);
    }
    __syncthreads();

    // FFN2 partial: acc2 += h_chunk @ W2[:, chunk]^T
#pragma unroll
    for (int i = 0; i < 2; ++i) {
      int t = wave + i * 8;
#pragma unroll
      for (int kb = 0; kb < 8; ++kb) {
        v16bf bf = load_b_packed(W2P, 32, t, c * 8 + kb);
#pragma unroll
        for (int mt = 0; mt < MT; ++mt)
          acc2[i][mt] =
              wmma_bf16(load_a_bf(hbuf, LDHC, mt * 16, kb * 32), bf, acc2[i][mt]);
      }
    }
    __syncthreads();  // before next chunk overwrites hbuf
  }

  // ---- 10. out = x + ffn + b2, store transposed --------------------------
#pragma unroll
  for (int i = 0; i < 2; ++i) {
    int t = wave + i * 8;
    int col = t * 16 + nn;
    float bias = b2[col];
    float* ocol = out + ((long)b * E + col) * N + n0;
#pragma unroll
    for (int mt = 0; mt < MT; ++mt)
#pragma unroll
      for (int r = 0; r < 8; ++r) {
        int m = mt * 16 + r + half * 8;
        ocol[m] = xf[m * LDX + col] + acc2[i][mt][r] + bias;
      }
  }
}

// ---------------------------------------------------------------------------
extern "C" void kernel_launch(void* const* d_in, const int* in_sizes, int n_in,
                              void* d_out, int out_size, void* d_ws, size_t ws_size,
                              hipStream_t stream) {
  (void)in_sizes; (void)n_in; (void)out_size; (void)ws_size;
  const float* img   = (const float*)d_in[0];
  const float* tgt   = (const float*)d_in[1];
  const float* bg    = (const float*)d_in[2];
  const float* Wp    = (const float*)d_in[3];
  const float* bp    = (const float*)d_in[4];
  const float* g_img = (const float*)d_in[5];
  const float* b_img = (const float*)d_in[6];
  const float* g_tok = (const float*)d_in[7];
  const float* b_tok = (const float*)d_in[8];
  const float* g_ffn = (const float*)d_in[9];
  const float* b_ffn = (const float*)d_in[10];
  const float* Wq    = (const float*)d_in[11];
  const float* Wk    = (const float*)d_in[12];
  const float* Wv    = (const float*)d_in[13];
  const float* bq    = (const float*)d_in[14];
  const float* bk    = (const float*)d_in[15];
  const float* bv    = (const float*)d_in[16];
  const float* Wo    = (const float*)d_in[17];
  const float* bo    = (const float*)d_in[18];
  const float* W1    = (const float*)d_in[19];
  const float* b1    = (const float*)d_in[20];
  const float* W2    = (const float*)d_in[21];
  const float* b2    = (const float*)d_in[22];

  // workspace layout
  float* Kbuf = (float*)d_ws;                 // [B,2,E]   = 8192 f32
  float* Vbuf = Kbuf + 16 * 2 * 256;          // [B,2,E]   = 8192 f32
  bf16*  pw   = (bf16*)(Vbuf + 16 * 2 * 256); // packed weights, bf16
  bf16*  WqP  = pw;                           //  16*8 *512 =  65536
  bf16*  WoP  = WqP + 65536;                  //  16*8 *512 =  65536
  bf16*  W1P  = WoP + 65536;                  //  64*8 *512 = 262144
  bf16*  W2P  = W1P + 262144;                 //  16*32*512 = 262144

  // pack weights into WMMA B-fragment bf16 layout (repacked every call:
  // deterministic, no cross-call state)
  pack_weight_kernel<<<16 * 8, 32, 0, stream>>>(Wq, WqP, 256);
  pack_weight_kernel<<<16 * 8, 32, 0, stream>>>(Wo, WoP, 256);
  pack_weight_kernel<<<64 * 8, 32, 0, stream>>>(W1, W1P, 256);
  pack_weight_kernel<<<16 * 32, 32, 0, stream>>>(W2, W2P, 1024);

  token_kv_kernel<<<32, 256, 0, stream>>>(tgt, bg, Wp, bp, g_tok, b_tok,
                                          Wk, Wv, bk, bv, Kbuf, Vbuf);

  // allow >64KB dynamic LDS (CDNA5: 320KB per WGP)
  (void)hipFuncSetAttribute((const void*)fused_block_kernel,
                            hipFuncAttributeMaxDynamicSharedMemorySize,
                            SMEM_BYTES);

  dim3 grid(4096 / M_ROWS, 16);  // (tiles per batch, B)
  fused_block_kernel<<<grid, 256, SMEM_BYTES, stream>>>(
      img, g_img, b_img, g_ffn, b_ffn, WqP, bq, WoP, bo,
      W1P, b1, W2P, b2, Kbuf, Vbuf, (float*)d_out);
}
